// GNNPolicy_57260503991113
// MI455X (gfx1250) — matrix operand order repaired and checked by reading, not compiled, not run on hardware
//
#include <hip/hip_runtime.h>

#define BB 64
#define VV 128
#define NNODE 8192           // BB*VV
#define EBUF 520192          // BB*VV*(VV-1)/2

typedef _Float16 f16;
typedef __attribute__((ext_vector_type(16))) _Float16 v16h;
typedef __attribute__((ext_vector_type(8)))  float    v8f;

union Frag { v16h h; int4 q[2]; };

// ---------------------------------------------------------------------------
// Generic WMMA GEMM:  C[m,n] = act( sum_k A[m,k]*Bw[n,k] + bias[n]
//                                   + addA[idxA[m],n] + addB[idxB[m],n]
//                                   + addG[(idxG?idxG[m]:m)>>gShift, n] ) * scl
// A: [M,K] f16 row-major (lda), Bw: [N,K] f16 row-major (ldb, weight layout).
// Block = 256 threads = 8 waves; wave w does output tile (blockIdx.x, y*8+w).
// ---------------------------------------------------------------------------
struct GemmP {
  const f16* A; int lda; long aStr;
  const f16* Bw; int ldb; long bStr;
  float* Cf; f16* Ch; int ldc; long cStr;
  int M, N, K;
  const float* bias;
  const float* addA; const int* idxA;
  const float* addB; const int* idxB;
  const float* addG; const int* idxG; int gShift;
  const int* pMv;            // dynamic valid-row count (null -> M)
  const float* scaleDiv;     // if non-null: multiply by 1/scaleDiv[0]
  int relu;
  int inplaceSync;           // barrier between reads and stores (in-place GEMM)
};

__global__ __launch_bounds__(256) void wmma_gemm_k(GemmP p)
{
  const int lane = threadIdx.x & 31;
  const int wave = threadIdx.x >> 5;
  const int m0   = blockIdx.x * 16;
  const int Mv   = p.pMv ? *p.pMv : p.M;
  const int n0   = (blockIdx.y * 8 + wave) * 16;
  const int row  = lane & 15;
  const int hi   = lane >> 4;
  const int bz   = blockIdx.z;

  const bool active = (m0 < Mv) && (n0 < p.N);
  v8f acc = {0.f,0.f,0.f,0.f,0.f,0.f,0.f,0.f};
  const f16* aRow = p.A  + (long)bz * p.aStr + (long)(m0 + row) * p.lda;
  const f16* bRow = p.Bw + (long)bz * p.bStr + (long)(n0 + row) * p.ldb;
  if (active) {
    for (int k = 0; k < p.K; k += 32) {
      Frag a, b;
      // 16-bit A 16x32 fragment: lane holds K = hi*8..+7 and 16+hi*8..+7
      a.q[0] = *(const int4*)(aRow + k + hi * 8);
      a.q[1] = *(const int4*)(aRow + k + 16 + hi * 8);
      // 16-bit B 32x16 fragment: lane = column, K = hi*8..+7 and 16+hi*8..+7
      b.q[0] = *(const int4*)(bRow + k + hi * 8);
      b.q[1] = *(const int4*)(bRow + k + 16 + hi * 8);
      acc = __builtin_amdgcn_wmma_f32_16x16x32_f16(false, a.h, false, b.h,
                                                   (short)0, acc, false, false);
    }
  }
  if (p.inplaceSync) __syncthreads();
  if (!active) return;

  const int n = n0 + row;
  const float bn  = p.bias ? p.bias[n] : 0.f;
  const float scl = p.scaleDiv ? 1.0f / p.scaleDiv[0] : 1.0f;
  #pragma unroll
  for (int v = 0; v < 8; ++v) {
    int m = m0 + v + 8 * hi;                 // C layout: M = v + 8*hi, N = lane&15
    if (m >= Mv) continue;
    float x = acc[v] + bn;
    if (p.addA) { int ia = p.idxA ? p.idxA[m] : m; x += p.addA[(long)ia * 128 + n]; }
    if (p.addB) { int ib = p.idxB ? p.idxB[m] : m; x += p.addB[(long)ib * 128 + n]; }
    if (p.addG) { int ig = p.idxG ? (p.idxG[m] >> p.gShift) : (m >> p.gShift);
                  x += p.addG[(long)ig * 128 + n]; }
    if (p.relu) x = fmaxf(x, 0.f);
    x *= scl;
    long co = (long)bz * p.cStr + (long)m * p.ldc + n;
    if (p.Cf) p.Cf[co] = x;
    if (p.Ch) p.Ch[co] = (f16)x;
  }
}

// ------------------------------- helpers -----------------------------------
__device__ __forceinline__ float redsum128(float v, float* sh)
{
  int t = threadIdx.x;
  sh[t] = v; __syncthreads();
  if (t < 64) sh[t] += sh[t + 64]; __syncthreads();
  if (t < 32) sh[t] += sh[t + 32]; __syncthreads();
  if (t < 16) sh[t] += sh[t + 16]; __syncthreads();
  if (t <  8) sh[t] += sh[t +  8]; __syncthreads();
  if (t <  4) sh[t] += sh[t +  4]; __syncthreads();
  if (t <  2) sh[t] += sh[t +  2]; __syncthreads();
  if (t <  1) sh[t] += sh[t +  1]; __syncthreads();
  float r = sh[0]; __syncthreads();
  return r;
}

// ------------------------ edge-list construction ---------------------------
__global__ void count_rows_k(const int* __restrict__ adj, int* __restrict__ cnt)
{
  int r = blockIdx.x * blockDim.x + threadIdx.x;   // b*128+i
  if (r >= NNODE) return;
  const int* p = adj + (long)r * 128;
  int c = 0;
  for (int j = 0; j < 128; ++j) c += p[j];
  cnt[r] = c;
}

__global__ void count_cols_k(const int* __restrict__ adj, int* __restrict__ cnt)
{
  int c = blockIdx.x * blockDim.x + threadIdx.x;   // b*128+j
  if (c >= NNODE) return;
  int b = c >> 7, j = c & 127;
  int s = 0;
  for (int i = 0; i < 128; ++i) s += adj[((long)(b * 128 + i)) * 128 + j];
  cnt[c] = s;
}

__global__ __launch_bounds__(1024) void scan8192_k(const int* __restrict__ cnt,
                                                   int* __restrict__ off,
                                                   int* __restrict__ pTot)
{
  __shared__ int part[1024];
  int t = threadIdx.x;
  int loc[8]; int s = 0;
  #pragma unroll
  for (int k = 0; k < 8; ++k) { loc[k] = s; s += cnt[t * 8 + k]; }
  part[t] = s; __syncthreads();
  for (int d = 1; d < 1024; d <<= 1) {
    int add = (t >= d) ? part[t - d] : 0;
    __syncthreads();
    part[t] += add;
    __syncthreads();
  }
  int base = (t == 0) ? 0 : part[t - 1];
  #pragma unroll
  for (int k = 0; k < 8; ++k) off[t * 8 + k] = base + loc[k];
  if (t == 1023) { off[8192] = part[1023]; if (pTot) *pTot = part[1023]; }
}

__global__ __launch_bounds__(128) void fill_edges_k(const int* __restrict__ adj,
    const int* __restrict__ rowOff, int* __restrict__ senders,
    int* __restrict__ receivers, int* __restrict__ edgeId)
{
  int r = blockIdx.x;                  // b*128+i
  int j = threadIdx.x;
  __shared__ int pre[128];
  int a = adj[(long)r * 128 + j];
  pre[j] = a; __syncthreads();
  for (int d = 1; d < 128; d <<= 1) {
    int add = (j >= d) ? pre[j - d] : 0;
    __syncthreads();
    pre[j] += add;
    __syncthreads();
  }
  int excl = pre[j] - a;
  if (a) {
    int idx = rowOff[r] + excl;
    senders[idx]   = r;
    receivers[idx] = ((r >> 7) << 7) + j;
    edgeId[(long)r * 128 + j] = idx;
  } else {
    edgeId[(long)r * 128 + j] = -1;
  }
}

__global__ __launch_bounds__(128) void fill_recv_k(const int* __restrict__ adj,
    const int* __restrict__ colOff, const int* __restrict__ edgeId,
    int* __restrict__ recvIdx)
{
  int c = blockIdx.x;                  // b*128+j
  int b = c >> 7, j = c & 127;
  int i = threadIdx.x;
  __shared__ int pre[128];
  int a = adj[((long)(b * 128 + i)) * 128 + j];
  pre[i] = a; __syncthreads();
  for (int d = 1; d < 128; d <<= 1) {
    int add = (i >= d) ? pre[i - d] : 0;
    __syncthreads();
    pre[i] += add;
    __syncthreads();
  }
  int excl = pre[i] - a;
  if (a) recvIdx[colOff[c] + excl] = edgeId[((long)(b * 128 + i)) * 128 + j];
}

// ------------------------------- init / cvt --------------------------------
__global__ void cvt_k(const float* __restrict__ s, f16* __restrict__ d, int n)
{
  int i = blockIdx.x * 256 + threadIdx.x;
  if (i < n) d[i] = (f16)s[i];
}

__global__ void init_edges_k(const float* __restrict__ emb, f16* __restrict__ e)
{
  long i = (long)blockIdx.x * 256 + threadIdx.x;
  if (i < (long)EBUF * 128) e[i] = (f16)emb[i & 127];
}

__global__ void init_nodes_k(const float* __restrict__ nemb, const float* __restrict__ gemb,
                             float* __restrict__ nodes, f16* __restrict__ nodesH,
                             float* __restrict__ glob,  f16* __restrict__ globH)
{
  int i = blockIdx.x * 256 + threadIdx.x;
  if (i < NNODE * 128) { float v = nemb[i & 16383]; nodes[i] = v; nodesH[i] = (f16)v; }
  if (i < 64 * 128)    { float g = gemb[i & 127];   glob[i]  = g; globH[i]  = (f16)g; }
}

__global__ void zero_f32_k(float* __restrict__ p, long n)
{
  long i = (long)blockIdx.x * 256 + threadIdx.x;
  if (i < n) p[i] = 0.f;
}

// ------------------------------ segment sums -------------------------------
__global__ __launch_bounds__(128) void segsum_rows_k(const f16* __restrict__ enew,
    const int* __restrict__ off, f16* __restrict__ out)
{
  int n = blockIdx.x, d = threadIdx.x;
  int s = off[n], e = off[n + 1];
  float acc = 0.f;
  for (int k = s; k < e; ++k) acc += (float)enew[(long)k * 128 + d];
  out[(long)n * 128 + d] = (f16)acc;
}

__global__ __launch_bounds__(128) void segsum_idx_k(const f16* __restrict__ enew,
    const int* __restrict__ off, const int* __restrict__ idxs, f16* __restrict__ out)
{
  int n = blockIdx.x, d = threadIdx.x;
  int s = off[n], e = off[n + 1];
  float acc = 0.f;
  for (int k = s; k < e; ++k) acc += (float)enew[(long)idxs[k] * 128 + d];
  out[(long)n * 128 + d] = (f16)acc;
}

__global__ __launch_bounds__(128) void build_gin_k(const f16* __restrict__ nnew,
    const f16* __restrict__ enew, const int* __restrict__ rowOff,
    const float* __restrict__ glob, f16* __restrict__ gin)
{
  int b = blockIdx.x, d = threadIdx.x;
  float ns = 0.f;
  for (int v = 0; v < 128; ++v) ns += (float)nnew[((long)(b * 128 + v)) * 128 + d];
  float es = 0.f;
  int e0 = rowOff[b * 128], e1 = rowOff[(b + 1) * 128];
  for (int k = e0; k < e1; ++k) es += (float)enew[(long)k * 128 + d];
  gin[(long)b * 384 + d]       = (f16)ns;
  gin[(long)b * 384 + 128 + d] = (f16)es;
  gin[(long)b * 384 + 256 + d] = (f16)glob[b * 128 + d];
}

// -------------------------------- layernorms -------------------------------
__global__ __launch_bounds__(128) void ln_edges_k(const int* __restrict__ pTot,
    f16* __restrict__ edges, const f16* __restrict__ enew,
    const float* __restrict__ w, const float* __restrict__ b)
{
  int e = blockIdx.x;
  if (e >= *pTot) return;
  int d = threadIdx.x;
  __shared__ float sh[128];
  long o = (long)e * 128 + d;
  float x = (float)edges[o] + (float)enew[o];
  float mean = redsum128(x, sh) * (1.f / 128.f);
  float dv = x - mean;
  float var = redsum128(dv * dv, sh) * (1.f / 128.f);
  edges[o] = (f16)(dv * rsqrtf(var + 1e-5f) * w[d] + b[d]);
}

__global__ __launch_bounds__(128) void ln_res_k(float* __restrict__ xf,
    const f16* __restrict__ add, f16* __restrict__ outH,
    const float* __restrict__ w, const float* __restrict__ b)
{
  int n = blockIdx.x, d = threadIdx.x;
  __shared__ float sh[128];
  long o = (long)n * 128 + d;
  float x = xf[o] + (float)add[o];
  float mean = redsum128(x, sh) * (1.f / 128.f);
  float dv = x - mean;
  float var = redsum128(dv * dv, sh) * (1.f / 128.f);
  float y = dv * rsqrtf(var + 1e-5f) * w[d] + b[d];
  xf[o] = y; outH[o] = (f16)y;
}

__global__ __launch_bounds__(128) void ln_f32h_k(const float* __restrict__ src,
    f16* __restrict__ dst, const float* __restrict__ w, const float* __restrict__ b)
{
  int n = blockIdx.x, d = threadIdx.x;
  __shared__ float sh[128];
  long o = (long)n * 128 + d;
  float x = src[o];
  float mean = redsum128(x, sh) * (1.f / 128.f);
  float dv = x - mean;
  float var = redsum128(dv * dv, sh) * (1.f / 128.f);
  dst[o] = (f16)(dv * rsqrtf(var + 1e-5f) * w[d] + b[d]);
}

// -------------------------------- attention --------------------------------
__global__ __launch_bounds__(128) void attn_k(const float* __restrict__ qh,
    const float* __restrict__ kh, const float* __restrict__ vh, f16* __restrict__ oH)
{
  int b = blockIdx.x, h = blockIdx.y, s = threadIdx.x;
  __shared__ float Ks[128][16];
  __shared__ float Vs[128][16];
  long base = ((long)(b * 128)) * 128 + h * 16;
  #pragma unroll
  for (int d = 0; d < 16; ++d) {
    Ks[s][d] = kh[base + (long)s * 128 + d];
    Vs[s][d] = vh[base + (long)s * 128 + d];
  }
  __syncthreads();
  float q[16];
  #pragma unroll
  for (int d = 0; d < 16; ++d) q[d] = qh[base + (long)s * 128 + d] * 0.25f;  // 1/sqrt(16)
  float mx = -3.0e38f;
  for (int t = 0; t < 128; ++t) {
    float sc = 0.f;
    #pragma unroll
    for (int d = 0; d < 16; ++d) sc += q[d] * Ks[t][d];
    mx = fmaxf(mx, sc);
  }
  float sum = 0.f, acc[16];
  #pragma unroll
  for (int d = 0; d < 16; ++d) acc[d] = 0.f;
  for (int t = 0; t < 128; ++t) {
    float sc = 0.f;
    #pragma unroll
    for (int d = 0; d < 16; ++d) sc += q[d] * Ks[t][d];
    float wgt = expf(sc - mx);
    sum += wgt;
    #pragma unroll
    for (int d = 0; d < 16; ++d) acc[d] += wgt * Vs[t][d];
  }
  float inv = 1.f / sum;
  #pragma unroll
  for (int d = 0; d < 16; ++d) oH[base + (long)s * 128 + d] = (f16)(acc[d] * inv);
}

// ---------------------------------- stop head ------------------------------
__global__ __launch_bounds__(128) void stop_k(const float* __restrict__ glob,
    const float* __restrict__ lnw, const float* __restrict__ lnb,
    const float* __restrict__ w0, const float* __restrict__ b0,
    const float* __restrict__ w1, const float* __restrict__ b1,
    const float* __restrict__ lnorm, float* __restrict__ out)
{
  int b = blockIdx.x, t = threadIdx.x;
  __shared__ float sh[128];
  __shared__ float gf[128];
  __shared__ float hid[128];
  float x = glob[b * 128 + t];
  float mean = redsum128(x, sh) * (1.f / 128.f);
  float dv = x - mean;
  float var = redsum128(dv * dv, sh) * (1.f / 128.f);
  gf[t] = dv * rsqrtf(var + 1e-5f) * lnw[t] + lnb[t];
  __syncthreads();
  float a = b0[t];
  for (int k = 0; k < 128; ++k) a += w0[t * 128 + k] * gf[k];
  hid[t] = fmaxf(a, 0.f);
  __syncthreads();
  float tot = redsum128(w1[t] * hid[t], sh);
  if (t == 0) out[(long)b * 16385 + 16384] = (tot + b1[0]) / lnorm[0];
}

// ============================ host orchestration ===========================
static inline void launch_gemm(hipStream_t st, const GemmP& p, int batch)
{
  dim3 g((unsigned)((p.M + 15) / 16), (unsigned)((p.N + 127) / 128), (unsigned)batch);
  wmma_gemm_k<<<g, dim3(256), 0, st>>>(p);
}

static inline GemmP mk(const f16* A, int lda, const f16* Bw, int ldb,
                       float* Cf, f16* Ch, int ldc, int M, int N, int K)
{
  GemmP p{};
  p.A = A; p.lda = lda; p.Bw = Bw; p.ldb = ldb;
  p.Cf = Cf; p.Ch = Ch; p.ldc = ldc;
  p.M = M; p.N = N; p.K = K; p.gShift = 7;
  return p;
}

extern "C" void kernel_launch(void* const* d_in, const int* in_sizes, int n_in,
                              void* d_out, int out_size, void* d_ws, size_t ws_size,
                              hipStream_t stream)
{
  (void)in_sizes; (void)n_in; (void)out_size; (void)ws_size;
  // ---- inputs (setup_inputs() dict order, depth-first) ----
  const int*   adj   = (const int*)  d_in[0];
  const float* nemb  = (const float*)d_in[1];
  const float* eemb  = (const float*)d_in[2];
  const float* gemb  = (const float*)d_in[3];
  const float* wn0   = (const float*)d_in[4];   const float* bn0 = (const float*)d_in[5];
  const float* wn1   = (const float*)d_in[6];   const float* bn1 = (const float*)d_in[7];
  const float* we0   = (const float*)d_in[8];   const float* be0 = (const float*)d_in[9];
  const float* we1   = (const float*)d_in[10];  const float* be1 = (const float*)d_in[11];
  const float* wg0   = (const float*)d_in[12];  const float* bg0 = (const float*)d_in[13];
  const float* wg1   = (const float*)d_in[14];  const float* bg1 = (const float*)d_in[15];
  const float* lnNw  = (const float*)d_in[16];  const float* lnNb = (const float*)d_in[17];
  const float* lnEw  = (const float*)d_in[18];  const float* lnEb = (const float*)d_in[19];
  const float* lnGw  = (const float*)d_in[20];  const float* lnGb = (const float*)d_in[21];
  const float* wpj   = (const float*)d_in[22];  const float* bpj  = (const float*)d_in[23];
  const float* wq    = (const float*)d_in[24];
  const float* wk    = (const float*)d_in[25];
  const float* wv    = (const float*)d_in[26];
  const float* wo    = (const float*)d_in[27];
  const float* lnPNw = (const float*)d_in[28];  const float* lnPNb = (const float*)d_in[29];
  const float* lnPGw = (const float*)d_in[30];  const float* lnPGb = (const float*)d_in[31];
  const float* ws0   = (const float*)d_in[32];  const float* bs0 = (const float*)d_in[33];
  const float* ws1   = (const float*)d_in[34];  const float* bs1 = (const float*)d_in[35];
  const float* wr0   = (const float*)d_in[36];  const float* br0 = (const float*)d_in[37];
  const float* wr1   = (const float*)d_in[38];  const float* br1 = (const float*)d_in[39];
  const float* wst0  = (const float*)d_in[40];  const float* bst0 = (const float*)d_in[41];
  const float* wst1  = (const float*)d_in[42];  const float* bst1 = (const float*)d_in[43];
  const float* lnorm = (const float*)d_in[44];
  float* out = (float*)d_out;

  // ---- workspace bump allocator ----
  char* base = (char*)d_ws;
  size_t off = 0;
  auto alloc = [&](size_t bytes) -> void* {
    off = (off + 255) & ~(size_t)255;
    void* p = base + off; off += bytes; return p;
  };
  const long NE = (long)EBUF * 128;
  f16*  h_edges = (f16*)alloc(NE * 2);            // edge features (f16 master)
  f16*  h_work  = (f16*)alloc(NE * 2);            // hidden, then e_new (in-place)
  float* f_nodes = (float*)alloc((size_t)NNODE * 128 * 4);
  f16*  h_nodes = (f16*)alloc((size_t)NNODE * 128 * 2);
  float* f_glob  = (float*)alloc(64 * 128 * 4);
  f16*  h_glob  = (f16*)alloc(64 * 128 * 2);
  int*  i_rowCnt = (int*)alloc(NNODE * 4);
  int*  i_rowOff = (int*)alloc((NNODE + 1) * 4);
  int*  i_colCnt = (int*)alloc(NNODE * 4);
  int*  i_colOff = (int*)alloc((NNODE + 1) * 4);
  int*  i_total  = (int*)alloc(4);
  int*  i_send   = (int*)alloc((size_t)EBUF * 4);
  int*  i_recv   = (int*)alloc((size_t)EBUF * 4);
  int*  i_rIdx   = (int*)alloc((size_t)EBUF * 4);
  int*  i_eId    = (int*)alloc((size_t)NNODE * 128 * 4);
  f16*  h_we0 = (f16*)alloc(65536 * 2);
  f16*  h_wn0 = (f16*)alloc(65536 * 2);
  f16*  h_wg0 = (f16*)alloc(49152 * 2);
  f16*  h_we1 = (f16*)alloc(16384 * 2);
  f16*  h_wn1 = (f16*)alloc(16384 * 2);
  f16*  h_wg1 = (f16*)alloc(16384 * 2);
  f16*  h_wpj = (f16*)alloc(49152 * 2);
  f16*  h_wq  = (f16*)alloc(16384 * 2);
  f16*  h_wk  = (f16*)alloc(16384 * 2);
  f16*  h_wv  = (f16*)alloc(16384 * 2);
  f16*  h_wo  = (f16*)alloc(16384 * 2);
  f16*  h_ws0 = (f16*)alloc(16384 * 2);
  f16*  h_ws1 = (f16*)alloc(16384 * 2);
  f16*  h_wr0 = (f16*)alloc(16384 * 2);
  f16*  h_wr1 = (f16*)alloc(16384 * 2);
  float* f_nsp  = (float*)alloc((size_t)NNODE * 128 * 4);
  float* f_nrp  = (float*)alloc((size_t)NNODE * 128 * 4);
  float* f_gpe  = (float*)alloc(64 * 128 * 4);
  float* f_gpn  = (float*)alloc(64 * 128 * 4);
  f16*  h_sent = (f16*)alloc((size_t)NNODE * 128 * 2);
  f16*  h_recv = (f16*)alloc((size_t)NNODE * 128 * 2);
  float* f_tmp1 = (float*)alloc((size_t)NNODE * 128 * 4);
  float* f_tmp2 = (float*)alloc((size_t)NNODE * 128 * 4);
  f16*  h_hn   = (f16*)alloc((size_t)NNODE * 128 * 2);
  f16*  h_nnew = (f16*)alloc((size_t)NNODE * 128 * 2);
  f16*  h_gin  = (f16*)alloc(64 * 384 * 2);
  f16*  h_hg   = (f16*)alloc(64 * 128 * 2);
  f16*  h_gnew = (f16*)alloc(64 * 128 * 2);
  f16*  h_nf   = (f16*)alloc((size_t)NNODE * 384 * 2);
  float* f_qh  = (float*)alloc((size_t)NNODE * 128 * 4);
  float* f_kh  = (float*)alloc((size_t)NNODE * 128 * 4);
  float* f_vh  = (float*)alloc((size_t)NNODE * 128 * 4);
  f16*  h_o    = (f16*)alloc((size_t)NNODE * 128 * 2);
  float* f_att = (float*)alloc((size_t)NNODE * 128 * 4);
  f16*  h_nfp  = (f16*)alloc((size_t)NNODE * 128 * 2);
  f16*  h_s    = (f16*)alloc((size_t)NNODE * 128 * 2);
  f16*  h_r    = (f16*)alloc((size_t)NNODE * 128 * 2);

  // ---- edge-list construction (deterministic, exact nonzero order) ----
  count_rows_k<<<32, 256, 0, stream>>>(adj, i_rowCnt);
  count_cols_k<<<32, 256, 0, stream>>>(adj, i_colCnt);
  scan8192_k<<<1, 1024, 0, stream>>>(i_rowCnt, i_rowOff, i_total);
  scan8192_k<<<1, 1024, 0, stream>>>(i_colCnt, i_colOff, nullptr);
  fill_edges_k<<<NNODE, 128, 0, stream>>>(adj, i_rowOff, i_send, i_recv, i_eId);
  fill_recv_k<<<NNODE, 128, 0, stream>>>(adj, i_colOff, i_eId, i_rIdx);

  // ---- init embeddings + convert weights to f16 ----
  init_edges_k<<<(unsigned)((NE + 255) / 256), 256, 0, stream>>>(eemb, h_edges);
  init_nodes_k<<<4096, 256, 0, stream>>>(nemb, gemb, f_nodes, h_nodes, f_glob, h_glob);
  auto cvt = [&](const float* s, f16* d, int n) {
    cvt_k<<<(n + 255) / 256, 256, 0, stream>>>(s, d, n);
  };
  cvt(we0, h_we0, 65536); cvt(wn0, h_wn0, 65536); cvt(wg0, h_wg0, 49152);
  cvt(we1, h_we1, 16384); cvt(wn1, h_wn1, 16384); cvt(wg1, h_wg1, 16384);
  cvt(wpj, h_wpj, 49152);
  cvt(wq, h_wq, 16384); cvt(wk, h_wk, 16384); cvt(wv, h_wv, 16384); cvt(wo, h_wo, 16384);
  cvt(ws0, h_ws0, 16384); cvt(ws1, h_ws1, 16384);
  cvt(wr0, h_wr0, 16384); cvt(wr1, h_wr1, 16384);

  // ---- 2 message-passing layers ----
  for (int layer = 0; layer < 2; ++layer) {
    // gprojE = glob @ we0[:,384:512]^T + be0 (fold bias)
    { GemmP p = mk(h_glob, 128, h_we0 + 384, 512, f_gpe, nullptr, 128, 64, 128, 128);
      p.bias = be0; launch_gemm(stream, p, 1); }
    // ns_proj / nr_proj = nodes @ we0[:,128:256]^T / we0[:,256:384]^T
    { GemmP p = mk(h_nodes, 128, h_we0 + 128, 512, f_nsp, nullptr, 128, NNODE, 128, 128);
      launch_gemm(stream, p, 1); }
    { GemmP p = mk(h_nodes, 128, h_we0 + 256, 512, f_nrp, nullptr, 128, NNODE, 128, 128);
      launch_gemm(stream, p, 1); }
    // edge hidden: h = relu(edges@we0a^T + ns[s] + nr[r] + gpe[b])
    { GemmP p = mk(h_edges, 128, h_we0, 512, nullptr, h_work, 128, EBUF, 128, 128);
      p.addA = f_nsp; p.idxA = i_send; p.addB = f_nrp; p.idxB = i_recv;
      p.addG = f_gpe; p.idxG = i_send; p.gShift = 7;
      p.pMv = i_total; p.relu = 1; launch_gemm(stream, p, 1); }
    // e_new = h @ we1^T + be1 (in-place over h_work)
    { GemmP p = mk(h_work, 128, h_we1, 128, nullptr, h_work, 128, EBUF, 128, 128);
      p.bias = be1; p.pMv = i_total; p.inplaceSync = 1; launch_gemm(stream, p, 1); }
    // deterministic segment sums (senders CSR contiguous; receivers via recvIdx)
    segsum_rows_k<<<NNODE, 128, 0, stream>>>(h_work, i_rowOff, h_sent);
    segsum_idx_k <<<NNODE, 128, 0, stream>>>(h_work, i_colOff, i_rIdx, h_recv);
    // node update
    { GemmP p = mk(h_glob, 128, h_wn0 + 384, 512, f_gpn, nullptr, 128, 64, 128, 128);
      p.bias = bn0; launch_gemm(stream, p, 1); }
    { GemmP p = mk(h_sent, 128, h_wn0 + 128, 512, f_tmp1, nullptr, 128, NNODE, 128, 128);
      launch_gemm(stream, p, 1); }
    { GemmP p = mk(h_recv, 128, h_wn0 + 256, 512, f_tmp2, nullptr, 128, NNODE, 128, 128);
      launch_gemm(stream, p, 1); }
    { GemmP p = mk(h_nodes, 128, h_wn0, 512, nullptr, h_hn, 128, NNODE, 128, 128);
      p.addA = f_tmp1; p.addB = f_tmp2; p.addG = f_gpn; p.gShift = 7;
      p.relu = 1; launch_gemm(stream, p, 1); }
    { GemmP p = mk(h_hn, 128, h_wn1, 128, nullptr, h_nnew, 128, NNODE, 128, 128);
      p.bias = bn1; launch_gemm(stream, p, 1); }
    // global update
    build_gin_k<<<64, 128, 0, stream>>>(h_nnew, h_work, i_rowOff, f_glob, h_gin);
    { GemmP p = mk(h_gin, 384, h_wg0, 384, nullptr, h_hg, 128, 64, 128, 384);
      p.bias = bg0; p.relu = 1; launch_gemm(stream, p, 1); }
    { GemmP p = mk(h_hg, 128, h_wg1, 128, nullptr, h_gnew, 128, 64, 128, 128);
      p.bias = bg1; launch_gemm(stream, p, 1); }
    // residual + layernorms
    ln_edges_k<<<EBUF, 128, 0, stream>>>(i_total, h_edges, h_work, lnEw, lnEb);
    ln_res_k<<<NNODE, 128, 0, stream>>>(f_nodes, h_nnew, h_nodes, lnNw, lnNb);
    ln_res_k<<<64, 128, 0, stream>>>(f_glob, h_gnew, h_glob, lnGw, lnGb);
  }

  // ---- attention head ----
  { GemmP p = mk(h_nodes, 128, h_wpj, 128, nullptr, h_nf, 384, NNODE, 384, 128);
    p.bias = bpj; launch_gemm(stream, p, 1); }                     // nf = [q|k|v]
  { GemmP p = mk(h_nf + 0,   384, h_wq, 128, f_qh, nullptr, 128, NNODE, 128, 128);
    launch_gemm(stream, p, 1); }
  { GemmP p = mk(h_nf + 128, 384, h_wk, 128, f_kh, nullptr, 128, NNODE, 128, 128);
    launch_gemm(stream, p, 1); }
  { GemmP p = mk(h_nf + 256, 384, h_wv, 128, f_vh, nullptr, 128, NNODE, 128, 128);
    launch_gemm(stream, p, 1); }
  attn_k<<<dim3(64, 8), 128, 0, stream>>>(f_qh, f_kh, f_vh, h_o);
  { GemmP p = mk(h_o, 128, h_wo, 128, f_att, nullptr, 128, NNODE, 128, 128);
    launch_gemm(stream, p, 1); }
  ln_f32h_k<<<NNODE, 128, 0, stream>>>(f_att, h_nfp, lnPNw, lnPNb);

  // ---- sender / receiver MLPs ----
  { GemmP p = mk(h_nfp, 128, h_ws0, 128, nullptr, h_hn, 128, NNODE, 128, 128);
    p.bias = bs0; p.relu = 1; launch_gemm(stream, p, 1); }
  { GemmP p = mk(h_hn, 128, h_ws1, 128, nullptr, h_s, 128, NNODE, 128, 128);
    p.bias = bs1; launch_gemm(stream, p, 1); }
  { GemmP p = mk(h_nfp, 128, h_wr0, 128, nullptr, h_hn, 128, NNODE, 128, 128);
    p.bias = br0; p.relu = 1; launch_gemm(stream, p, 1); }
  { GemmP p = mk(h_hn, 128, h_wr1, 128, nullptr, h_r, 128, NNODE, 128, 128);
    p.bias = br1; launch_gemm(stream, p, 1); }

  // ---- edge logits: per-batch s_b @ r_b^T / logits_norm -> d_out ----
  { GemmP p = mk(h_s, 128, h_r, 128, out, nullptr, 128, 128, 128, 128);
    p.aStr = 128 * 128; p.bStr = 128 * 128; p.cStr = 16385;
    p.scaleDiv = lnorm; launch_gemm(stream, p, 64); }

  // ---- stop logit (incl. ln_post_globals) + zero bwd output ----
  stop_k<<<64, 128, 0, stream>>>(f_glob, lnPGw, lnPGb, wst0, bst0, wst1, bst1, lnorm, out);
  zero_f32_k<<<4096, 256, 0, stream>>>(out + (long)64 * 16385, (long)64 * 16384);
}
